// EnzymeCompoundCrossAttention_86655260164933
// MI455X (gfx1250) — matrix operand order repaired and checked by reading, not compiled, not run on hardware
//
#include <hip/hip_runtime.h>

// ---------------------------------------------------------------------------
// MI455X (gfx1250) implementation of EnzymeCompoundCrossAttention.
// Compute-bound (~150 GFLOP vs ~8us of HBM traffic at 23.3 TB/s), so all
// matmuls run on v_wmma_f32_16x16x32_bf16 (bf16 inputs, fp32 accumulate).
// Softmax / bias / mask / means stay fp32. wave32 everywhere.
// Round 2: 32x64 per-wave tiles in the projection GEMM (B-fragment reused 2x)
// and software-pipelined (double-buffered) fragment loads so global_load_b128
// for step i+1 issues under the WMMA of step i instead of s_wait_loadcnt 0x0
// before every matrix op.
// ---------------------------------------------------------------------------

typedef __attribute__((ext_vector_type(16))) __bf16 v16bf;
typedef __attribute__((ext_vector_type(8)))  float  v8f;
typedef __attribute__((ext_vector_type(4)))  float  f32x4;

union FragAB {
  v16bf v;
  f32x4 h[2];
};

__device__ __forceinline__ unsigned short f32_to_bf16(float f) {
  unsigned int u = __float_as_uint(f);
  u += 0x7FFFu + ((u >> 16) & 1u);   // round-to-nearest-even
  return (unsigned short)(u >> 16);
}

__device__ __forceinline__ f32x4 ld16B(const unsigned short* p) {
  return *reinterpret_cast<const f32x4*>(p);
}

__device__ __forceinline__ v8f wmma_bf16(const FragAB& a, const FragAB& b, v8f c) {
  return __builtin_amdgcn_wmma_f32_16x16x32_bf16(false, a.v, false, b.v,
                                                 (short)0, c, false, false);
}

// ---------------------------------------------------------------------------
// fp32 -> bf16 elementwise
__global__ __launch_bounds__(256)
void k_cvt_bf16(const float* __restrict__ src, unsigned short* __restrict__ dst, int n) {
  int i = blockIdx.x * blockDim.x + threadIdx.x;
  if (i < n) dst[i] = f32_to_bf16(src[i]);
}

// fp32 W[K][N] -> bf16 Wt[N][K]  (so WMMA B-fragments are contiguous)
__global__ __launch_bounds__(256)
void k_wt_transpose(const float* __restrict__ W, unsigned short* __restrict__ Wt,
                    int K, int N) {
  int i = blockIdx.x * blockDim.x + threadIdx.x;
  if (i < K * N) {
    int k = i / N, n = i - k * N;
    Wt[(size_t)n * K + k] = f32_to_bf16(W[i]);
  }
}

// ---------------------------------------------------------------------------
// Projection GEMM:  out(bf16)[M][512] = A(bf16)[M][Kd] @ Wt(bf16)[512][Kd]^T + bias
// grid.x = M/32 ; 256 threads = 8 waves ; wave w computes a 32(M) x 64(N) tile
// (cols [w*64, w*64+64)), so each B fragment is reused by two WMMAs.
// storeT=1: store transposed per batch of L rows -> out[b][n][l] (for V tensors).
__global__ __launch_bounds__(256)
void k_proj_gemm(const unsigned short* __restrict__ A,
                 const unsigned short* __restrict__ Wt,
                 const float* __restrict__ bias,
                 unsigned short* __restrict__ out,
                 int Kd, int L, int storeT) {
  const int N = 512;
  int mt   = blockIdx.x;
  int wid  = threadIdx.x >> 5;
  int lane = threadIdx.x & 31;
  int lm   = lane & 15;
  int hi   = lane >> 4;

  v8f acc0[4], acc1[4];
#pragma unroll
  for (int f = 0; f < 4; ++f)
#pragma unroll
    for (int r = 0; r < 8; ++r) { acc0[f][r] = 0.0f; acc1[f][r] = 0.0f; }

  const unsigned short* Arow0 = A + (size_t)(mt * 32 + lm) * Kd + hi * 8;
  const unsigned short* Arow1 = Arow0 + (size_t)16 * Kd;
  const unsigned short* Bbase = Wt + (size_t)(wid * 64 + lm) * Kd + hi * 8;

  FragAB a0, a1, b, bn;
  a0.h[0] = ld16B(Arow0);      a0.h[1] = ld16B(Arow0 + 16);
  a1.h[0] = ld16B(Arow1);      a1.h[1] = ld16B(Arow1 + 16);
  b.h[0]  = ld16B(Bbase);      b.h[1]  = ld16B(Bbase + 16);
  bn = b;

  for (int kk = 0; kk < Kd; kk += 32) {
    bool last = (kk + 32 >= Kd);
    FragAB a0n = a0, a1n = a1;
#pragma unroll
    for (int f = 0; f < 4; ++f) {
      if (f < 3) {                       // prefetch next B fragment (same k-step)
        const unsigned short* Bp = Bbase + (size_t)(f + 1) * 16 * Kd + kk;
        bn.h[0] = ld16B(Bp);
        bn.h[1] = ld16B(Bp + 16);
      } else if (!last) {                // prefetch first B + both A of next k-step
        const unsigned short* Bp = Bbase + kk + 32;
        bn.h[0]  = ld16B(Bp);
        bn.h[1]  = ld16B(Bp + 16);
        a0n.h[0] = ld16B(Arow0 + kk + 32);
        a0n.h[1] = ld16B(Arow0 + kk + 48);
        a1n.h[0] = ld16B(Arow1 + kk + 32);
        a1n.h[1] = ld16B(Arow1 + kk + 48);
      }
      acc0[f] = wmma_bf16(a0, b, acc0[f]);
      acc1[f] = wmma_bf16(a1, b, acc1[f]);
      b = bn;
    }
    a0 = a0n;
    a1 = a1n;
  }

#pragma unroll
  for (int f = 0; f < 4; ++f) {
    int n = wid * 64 + f * 16 + lm;
    float bv = bias[n];
#pragma unroll
    for (int half = 0; half < 2; ++half) {
      v8f& ac = half ? acc1[f] : acc0[f];
#pragma unroll
      for (int r = 0; r < 8; ++r) {
        int m = mt * 32 + half * 16 + r + hi * 8;
        unsigned short val = f32_to_bf16(ac[r] + bv);
        if (!storeT) {
          out[(size_t)m * N + n] = val;
        } else {
          int bb = m / L;
          int l  = m - bb * L;
          out[((size_t)bb * N + n) * L + l] = val;   // [b][n][l]
        }
      }
    }
  }
}

// ---------------------------------------------------------------------------
// Fused cross-attention for one (batch, 16-query tile).
// Q: bf16 [B*Lq][512] ; K: bf16 [B*LK][512] ; Vt: bf16 [B][512][LK]
// biasMode: 0 none | 1 bias[b][n][q] (interaction^T) | 2 bias[b][q][n]
// O: fp32 [B*Lq][512]
template <int LK>
__global__ __launch_bounds__(256)
void k_attn(const unsigned short* __restrict__ Q,
            const unsigned short* __restrict__ K,
            const unsigned short* __restrict__ Vt,
            const unsigned char* __restrict__ qmask,
            const unsigned char* __restrict__ kmask,
            const float* __restrict__ bias,
            int biasMode, int Lq,
            float* __restrict__ O) {
  const int D = 512;
  constexpr int COLS  = LK / 8;    // key columns per wave
  constexpr int NFRAG = COLS / 16; // 16x16 fragments per wave
  __shared__ __align__(16) float          Sl[16 * LK];
  __shared__ __align__(16) unsigned short Pl[16 * LK];

  int nQt  = Lq >> 4;
  int b    = blockIdx.x / nQt;
  int qt   = blockIdx.x - b * nQt;
  int wid  = threadIdx.x >> 5;
  int lane = threadIdx.x & 31;
  int lm   = lane & 15;
  int hi   = lane >> 4;

  // ---- Phase 1: S = (Q K^T) * 1/sqrt(512) + bias, masked ----
  {
    v8f acc[NFRAG];
#pragma unroll
    for (int f = 0; f < NFRAG; ++f)
#pragma unroll
      for (int r = 0; r < 8; ++r) acc[f][r] = 0.0f;

    const unsigned short* Qrow  = Q + (size_t)(b * Lq + qt * 16 + lm) * D + hi * 8;
    const unsigned short* Bbase = K + (size_t)(b * LK + wid * COLS + lm) * D + hi * 8;

    FragAB a, bb, bn;
    a.h[0]  = ld16B(Qrow);       a.h[1]  = ld16B(Qrow + 16);
    bb.h[0] = ld16B(Bbase);      bb.h[1] = ld16B(Bbase + 16);
    bn = bb;

    for (int kk = 0; kk < D; kk += 32) {
      bool last = (kk + 32 >= D);
      FragAB an = a;
#pragma unroll
      for (int f = 0; f < NFRAG; ++f) {
        if (f + 1 < NFRAG) {
          const unsigned short* Bp = Bbase + (size_t)(f + 1) * 16 * D + kk;
          bn.h[0] = ld16B(Bp);
          bn.h[1] = ld16B(Bp + 16);
        } else if (!last) {
          const unsigned short* Bp = Bbase + kk + 32;
          bn.h[0] = ld16B(Bp);
          bn.h[1] = ld16B(Bp + 16);
          an.h[0] = ld16B(Qrow + kk + 32);
          an.h[1] = ld16B(Qrow + kk + 48);
        }
        acc[f] = wmma_bf16(a, bb, acc[f]);
        bb = bn;
      }
      a = an;
    }

    const float scale = 0.04419417382415922f;  // 1/sqrt(512)
#pragma unroll
    for (int f = 0; f < NFRAG; ++f) {
      int n = wid * COLS + f * 16 + lm;
#pragma unroll
      for (int r = 0; r < 8; ++r) {
        int m = r + hi * 8;
        int q = qt * 16 + m;
        float s = acc[f][r] * scale;
        if (biasMode == 1)      s += bias[((size_t)b * LK + n) * Lq + q];
        else if (biasMode == 2) s += bias[((size_t)b * Lq + q) * LK + n];
        bool ok = (qmask[(size_t)b * Lq + q] != 0) && (kmask[(size_t)b * LK + n] != 0);
        Sl[m * LK + n] = ok ? s : -1.0e9f;
      }
    }
  }
  __syncthreads();

  // ---- Phase 2: row softmax (one wave per row, wave32 shuffle reductions) ----
  for (int r = wid; r < 16; r += 8) {
    float mx = -3.0e38f;
    for (int k = lane; k < LK; k += 32) mx = fmaxf(mx, Sl[r * LK + k]);
#pragma unroll
    for (int off = 16; off >= 1; off >>= 1) mx = fmaxf(mx, __shfl_xor(mx, off, 32));
    float sum = 0.0f;
    for (int k = lane; k < LK; k += 32) {
      float e = __expf(Sl[r * LK + k] - mx);
      Sl[r * LK + k] = e;
      sum += e;
    }
#pragma unroll
    for (int off = 16; off >= 1; off >>= 1) sum += __shfl_xor(sum, off, 32);
    float inv = 1.0f / sum;
    for (int k = lane; k < LK; k += 32) Pl[r * LK + k] = f32_to_bf16(Sl[r * LK + k] * inv);
  }
  __syncthreads();

  // ---- Phase 3: O = P @ V  (P from LDS via ds_load, Vt contiguous per dim) ----
  {
    v8f acc[4];
#pragma unroll
    for (int f = 0; f < 4; ++f)
#pragma unroll
      for (int r = 0; r < 8; ++r) acc[f][r] = 0.0f;

    const unsigned short* Prow = &Pl[lm * LK + hi * 8];
    const unsigned short* Bb   = Vt + (size_t)(b * D + wid * 64 + lm) * LK + hi * 8;

    FragAB a, bb, bn;
    a.h[0]  = ld16B(Prow);       a.h[1]  = ld16B(Prow + 16);
    bb.h[0] = ld16B(Bb);         bb.h[1] = ld16B(Bb + 16);
    bn = bb;

    for (int kk = 0; kk < LK; kk += 32) {
      bool last = (kk + 32 >= LK);
      FragAB an = a;
#pragma unroll
      for (int f = 0; f < 4; ++f) {
        if (f < 3) {
          const unsigned short* Bp = Bb + (size_t)(f + 1) * 16 * LK + kk;
          bn.h[0] = ld16B(Bp);
          bn.h[1] = ld16B(Bp + 16);
        } else if (!last) {
          const unsigned short* Bp = Bb + kk + 32;
          bn.h[0] = ld16B(Bp);
          bn.h[1] = ld16B(Bp + 16);
          an.h[0] = ld16B(Prow + kk + 32);
          an.h[1] = ld16B(Prow + kk + 48);
        }
        acc[f] = wmma_bf16(a, bb, acc[f]);
        bb = bn;
      }
      a = an;
    }

#pragma unroll
    for (int f = 0; f < 4; ++f) {
      int d = wid * 64 + f * 16 + lm;
#pragma unroll
      for (int r = 0; r < 8; ++r) {
        int m = r + hi * 8;
        O[(size_t)(b * Lq + qt * 16 + m) * D + d] = acc[f][r];
      }
    }
  }
}

// ---------------------------------------------------------------------------
// Mean over queries into d_out[b][sec*512 + d]
__global__ __launch_bounds__(256)
void k_mean(const float* __restrict__ O, float* __restrict__ out, int Lq, int sec) {
  int idx = blockIdx.x * blockDim.x + threadIdx.x;
  if (idx >= 32 * 512) return;
  int b = idx >> 9;
  int d = idx & 511;
  const float* p = O + ((size_t)b * Lq) * 512 + d;
  float s = 0.0f;
  for (int q = 0; q < Lq; ++q) s += p[(size_t)q * 512];
  out[b * 2048 + sec * 512 + d] = s * (1.0f / (float)Lq);
}

// ---------------------------------------------------------------------------
extern "C" void kernel_launch(void* const* d_in, const int* in_sizes, int n_in,
                              void* d_out, int out_size, void* d_ws, size_t ws_size,
                              hipStream_t stream) {
  (void)in_sizes; (void)n_in; (void)out_size; (void)ws_size;
  constexpr int B = 32, LE = 512, LC = 128, DE = 1280, DC = 256, DO = 512;

  const float* enz  = (const float*)d_in[0];
  const float* sub  = (const float*)d_in[1];
  const float* prod = (const float*)d_in[2];
  const unsigned char* enz_m  = (const unsigned char*)d_in[3];
  const unsigned char* sub_m  = (const unsigned char*)d_in[4];
  const unsigned char* prod_m = (const unsigned char*)d_in[5];
  const float* iw   = (const float*)d_in[6];
  const float* Wq_e = (const float*)d_in[7];  const float* bq_e = (const float*)d_in[8];
  const float* Wk_e = (const float*)d_in[9];  const float* bk_e = (const float*)d_in[10];
  const float* Wv_e = (const float*)d_in[11]; const float* bv_e = (const float*)d_in[12];
  const float* Wq_s = (const float*)d_in[13]; const float* bq_s = (const float*)d_in[14];
  const float* Wk_s = (const float*)d_in[15]; const float* bk_s = (const float*)d_in[16];
  const float* Wv_s = (const float*)d_in[17]; const float* bv_s = (const float*)d_in[18];
  const float* Wq_p = (const float*)d_in[19]; const float* bq_p = (const float*)d_in[20];
  const float* Wk_p = (const float*)d_in[21]; const float* bk_p = (const float*)d_in[22];
  const float* Wv_p = (const float*)d_in[23]; const float* bv_p = (const float*)d_in[24];
  float* out = (float*)d_out;

  char* wsp = (char*)d_ws;
  size_t off = 0;
  auto alloc_us = [&](size_t elems) -> unsigned short* {
    unsigned short* p = (unsigned short*)(wsp + off);
    off = (off + elems * 2 + 255) & ~(size_t)255;
    return p;
  };
  auto alloc_f = [&](size_t elems) -> float* {
    float* p = (float*)(wsp + off);
    off = (off + elems * 4 + 255) & ~(size_t)255;
    return p;
  };

  // bf16 feature copies
  unsigned short* enz_bf  = alloc_us((size_t)B * LE * DE);
  unsigned short* sub_bf  = alloc_us((size_t)B * LC * DC);
  unsigned short* prod_bf = alloc_us((size_t)B * LC * DC);
  // bf16 transposed weights [N=512][K]
  unsigned short* WqeT = alloc_us((size_t)DO * DE);
  unsigned short* WkeT = alloc_us((size_t)DO * DC);
  unsigned short* WveT = alloc_us((size_t)DO * DC);
  unsigned short* WqsT = alloc_us((size_t)DO * DC);
  unsigned short* WksT = alloc_us((size_t)DO * DE);
  unsigned short* WvsT = alloc_us((size_t)DO * DE);
  unsigned short* WqpT = alloc_us((size_t)DO * DC);
  unsigned short* WkpT = alloc_us((size_t)DO * DE);
  unsigned short* WvpT = alloc_us((size_t)DO * DE);
  // projected tensors (bf16); V tensors stored transposed per batch [D][Lk]
  unsigned short* Qenz    = alloc_us((size_t)B * LE * DO);
  unsigned short* Ksub    = alloc_us((size_t)B * LC * DO);
  unsigned short* VsubT   = alloc_us((size_t)B * DO * LC);
  unsigned short* Kprod   = alloc_us((size_t)B * LC * DO);
  unsigned short* VprodT  = alloc_us((size_t)B * DO * LC);
  unsigned short* Qsub    = alloc_us((size_t)B * LC * DO);
  unsigned short* Qprod   = alloc_us((size_t)B * LC * DO);
  unsigned short* Kenz_s  = alloc_us((size_t)B * LE * DO);
  unsigned short* VenzT_s = alloc_us((size_t)B * DO * LE);
  unsigned short* Kenz_p  = alloc_us((size_t)B * LE * DO);
  unsigned short* VenzT_p = alloc_us((size_t)B * DO * LE);
  // attention output scratch (reused by all four attentions)
  float* O = alloc_f((size_t)B * LE * DO);

  const int TPB = 256;
  auto g1 = [](size_t n) { return (unsigned)((n + 255) / 256); };

  // --- conversions ---
  k_cvt_bf16<<<g1((size_t)B * LE * DE), TPB, 0, stream>>>(enz, enz_bf, B * LE * DE);
  k_cvt_bf16<<<g1((size_t)B * LC * DC), TPB, 0, stream>>>(sub, sub_bf, B * LC * DC);
  k_cvt_bf16<<<g1((size_t)B * LC * DC), TPB, 0, stream>>>(prod, prod_bf, B * LC * DC);
  k_wt_transpose<<<g1((size_t)DE * DO), TPB, 0, stream>>>(Wq_e, WqeT, DE, DO);
  k_wt_transpose<<<g1((size_t)DC * DO), TPB, 0, stream>>>(Wk_e, WkeT, DC, DO);
  k_wt_transpose<<<g1((size_t)DC * DO), TPB, 0, stream>>>(Wv_e, WveT, DC, DO);
  k_wt_transpose<<<g1((size_t)DC * DO), TPB, 0, stream>>>(Wq_s, WqsT, DC, DO);
  k_wt_transpose<<<g1((size_t)DE * DO), TPB, 0, stream>>>(Wk_s, WksT, DE, DO);
  k_wt_transpose<<<g1((size_t)DE * DO), TPB, 0, stream>>>(Wv_s, WvsT, DE, DO);
  k_wt_transpose<<<g1((size_t)DC * DO), TPB, 0, stream>>>(Wq_p, WqpT, DC, DO);
  k_wt_transpose<<<g1((size_t)DE * DO), TPB, 0, stream>>>(Wk_p, WkpT, DE, DO);
  k_wt_transpose<<<g1((size_t)DE * DO), TPB, 0, stream>>>(Wv_p, WvpT, DE, DO);

  // --- projections (WMMA GEMMs, 32-row tiles) ---
  const int ME = B * LE / 32, MC = B * LC / 32;
  k_proj_gemm<<<ME, TPB, 0, stream>>>(enz_bf,  WqeT, bq_e, Qenz,    DE, LE, 0);
  k_proj_gemm<<<MC, TPB, 0, stream>>>(sub_bf,  WkeT, bk_e, Ksub,    DC, LC, 0);
  k_proj_gemm<<<MC, TPB, 0, stream>>>(sub_bf,  WveT, bv_e, VsubT,   DC, LC, 1);
  k_proj_gemm<<<MC, TPB, 0, stream>>>(prod_bf, WkeT, bk_e, Kprod,   DC, LC, 0);
  k_proj_gemm<<<MC, TPB, 0, stream>>>(prod_bf, WveT, bv_e, VprodT,  DC, LC, 1);
  k_proj_gemm<<<MC, TPB, 0, stream>>>(sub_bf,  WqsT, bq_s, Qsub,    DC, LC, 0);
  k_proj_gemm<<<ME, TPB, 0, stream>>>(enz_bf,  WksT, bk_s, Kenz_s,  DE, LE, 0);
  k_proj_gemm<<<ME, TPB, 0, stream>>>(enz_bf,  WvsT, bv_s, VenzT_s, DE, LE, 1);
  k_proj_gemm<<<MC, TPB, 0, stream>>>(prod_bf, WqpT, bq_p, Qprod,   DC, LC, 0);
  k_proj_gemm<<<ME, TPB, 0, stream>>>(enz_bf,  WkpT, bk_p, Kenz_p,  DE, LE, 0);
  k_proj_gemm<<<ME, TPB, 0, stream>>>(enz_bf,  WvpT, bv_p, VenzT_p, DE, LE, 1);

  // --- attentions + means (O reused sequentially on the stream) ---
  // 1) enzyme_subs: Q=enz, K/V=sub, bias = interaction^T  -> section 0
  k_attn<LC><<<B * (LE / 16), TPB, 0, stream>>>(Qenz, Ksub, VsubT, enz_m, sub_m,
                                                iw, 1, LE, O);
  k_mean<<<g1((size_t)B * DO), TPB, 0, stream>>>(O, out, LE, 0);
  // 2) subs_enzyme: Q=sub, K/V=enz, bias = interaction    -> section 1
  k_attn<LE><<<B * (LC / 16), TPB, 0, stream>>>(Qsub, Kenz_s, VenzT_s, sub_m, enz_m,
                                                iw, 2, LC, O);
  k_mean<<<g1((size_t)B * DO), TPB, 0, stream>>>(O, out, LC, 1);
  // 3) enzyme_prod: Q=enz (reused), K/V=prod, no bias     -> section 2
  k_attn<LC><<<B * (LE / 16), TPB, 0, stream>>>(Qenz, Kprod, VprodT, enz_m, prod_m,
                                                nullptr, 0, LE, O);
  k_mean<<<g1((size_t)B * DO), TPB, 0, stream>>>(O, out, LE, 2);
  // 4) prod_enzyme: Q=prod, K/V=enz, no bias              -> section 3
  k_attn<LE><<<B * (LC / 16), TPB, 0, stream>>>(Qprod, Kenz_p, VenzT_p, prod_m, enz_m,
                                                nullptr, 0, LC, O);
  k_mean<<<g1((size_t)B * DO), TPB, 0, stream>>>(O, out, LC, 3);
}
